// EpisodicEchoHead_75892072121081
// MI455X (gfx1250) — matrix operand order repaired and checked by reading, not compiled, not run on hardware
//
#include <hip/hip_runtime.h>
#include <math.h>

// EpisodicEchoHead fused attention for gfx1250 (MI455X).
// HBM-bound problem (512MB history, ~0.5 GFLOP) -> single-pass flash-style
// softmax-attention; v_wmma_f32_16x16x32_bf16 computes the 16-row score tiles.

#define B_       16
#define D_       2048
#define TWO_D    4096
#define H_       2048
#define SEGS     16          // H segments per batch
#define ROWS_WG  128         // H_/SEGS rows per workgroup (8 waves x 16 rows)
#define PSTRIDE  4112        // partial record: [0]=M, [1]=L, [16..4111]=acc
#define LUT_N    4096

typedef __attribute__((ext_vector_type(16))) __bf16 v16bf;
typedef __attribute__((ext_vector_type(8)))  float  v8f;

__device__ __forceinline__ float lut_cos(float theta) {
  // idx = round(theta * N / 2pi) mod N ; RNE rint matches jnp.round
  float fi = rintf(theta * 651.8986469044033f);        // 4096 / (2*pi)
  int idx = ((int)fi) & (LUT_N - 1);                    // power-of-2 mod, sign-safe
  return __cosf((float)idx * 0.0015339807878856412f);   // 2*pi / 4096
}

// ---------------------------------------------------------------- kernel 1
__global__ void __launch_bounds__(256)
build_query(const float* __restrict__ cr, const float* __restrict__ ci,
            const float* __restrict__ wq, const float* __restrict__ bq,
            const float* __restrict__ t_p, __bf16* __restrict__ qout) {
  int g = blockIdx.x * 256 + threadIdx.x;      // 0 .. B*2D-1
  int b = g >> 12;
  int j = g & (TWO_D - 1);
  int jj = (j < D_) ? j : (j - D_);
  float x  = (j < D_) ? cr[b * D_ + jj] : ci[b * D_ + jj];
  float wl = 1.0f + fabsf(wq[jj]);
  float th = x / wl + bq[jj] + t_p[0] * 1.6180339887498949f;  // + t*PHI
  qout[g] = (__bf16)lut_cos(th);
}

// ---------------------------------------------------------------- kernel 2
__global__ void __launch_bounds__(256)
flash_partial(const float* __restrict__ hr, const float* __restrict__ hgi,
              const __bf16* __restrict__ qbf, float* __restrict__ partial) {
  __shared__ float s_m[8];
  __shared__ float s_l[8];
  __shared__ float s_acc[TWO_D];

  const int b    = blockIdx.x / SEGS;
  const int seg  = blockIdx.x % SEGS;
  const int tid  = threadIdx.x;
  const int wave = tid >> 5;
  const int lane = tid & 31;
  const int mrow = lane & 15;       // A-matrix row this lane feeds
  const int hi16 = lane >> 4;       // lane half (K sub-block select)

  const int h0 = seg * ROWS_WG + wave * 16;          // this wave's 16-row tile
  const float* rowR = hr  + ((size_t)b * H_ + h0 + mrow) * D_;
  const float* rowI = hgi + ((size_t)b * H_ + h0 + mrow) * D_;
  const __bf16* q   = qbf + (size_t)b * TWO_D;

  // ---- Phase A: scores for 16 rows via WMMA, K = 0..4095 in chunks of 32.
  // A layout (16-bit 16x32): lane holds K = hi16*16+{0..15}.
  // B layout (16-bit 32x16): lane holds K = hi16*16+{0..15}; query broadcast
  // into all 16 columns -> every D column equals the score vector.
  v8f acc = {0.f, 0.f, 0.f, 0.f, 0.f, 0.f, 0.f, 0.f};
#pragma unroll 1
  for (int c = 0; c < 128; ++c) {
    const float* rp = (c < 64) ? rowR : rowI;        // uniform select
    const int k0 = (c & 63) * 32 + hi16 * 16;
    float4 a0 = *(const float4*)(rp + k0);
    float4 a1 = *(const float4*)(rp + k0 + 4);
    float4 a2 = *(const float4*)(rp + k0 + 8);
    float4 a3 = *(const float4*)(rp + k0 + 12);
    v16bf A;
    A[0]  = (__bf16)a0.x; A[1]  = (__bf16)a0.y; A[2]  = (__bf16)a0.z; A[3]  = (__bf16)a0.w;
    A[4]  = (__bf16)a1.x; A[5]  = (__bf16)a1.y; A[6]  = (__bf16)a1.z; A[7]  = (__bf16)a1.w;
    A[8]  = (__bf16)a2.x; A[9]  = (__bf16)a2.y; A[10] = (__bf16)a2.z; A[11] = (__bf16)a2.w;
    A[12] = (__bf16)a3.x; A[13] = (__bf16)a3.y; A[14] = (__bf16)a3.z; A[15] = (__bf16)a3.w;
    v16bf Bq = *(const v16bf*)(q + c * 32 + hi16 * 16);
    acc = __builtin_amdgcn_wmma_f32_16x16x32_bf16(
        false, A, false, Bq, (short)0, acc, false, false);
  }

  // c[r]: lanes 0-15 hold rows h0+r, lanes 16-31 hold rows h0+8+r.
  const float invscale = 1.0f / 64.0f;               // 1/sqrt(2D)
  float s[8], w[8];
  float mw = -1e30f;
#pragma unroll
  for (int r = 0; r < 8; ++r) { s[r] = acc[r] * invscale; mw = fmaxf(mw, s[r]); }
  mw = fmaxf(mw, __shfl_xor(mw, 16, 32));            // max over all 16 rows
  float es = 0.f;
#pragma unroll
  for (int r = 0; r < 8; ++r) { w[r] = __expf(s[r] - mw); es += w[r]; }
  float lw = es + __shfl_xor(es, 16, 32);            // sum-exp over 16 rows

  // All 16 row-weights in every lane, indexed by absolute tile row.
  float wr[16];
#pragma unroll
  for (int r = 0; r < 8; ++r) {
    float o = __shfl_xor(w[r], 16, 32);
    wr[r]     = hi16 ? o    : w[r];
    wr[8 + r] = hi16 ? w[r] : o;
  }

  if (lane == 0) { s_m[wave] = mw; s_l[wave] = lw; }
  for (int k = tid; k < TWO_D; k += 256) s_acc[k] = 0.f;
  __syncthreads();

  float M = s_m[0];
#pragma unroll
  for (int wv = 1; wv < 8; ++wv) M = fmaxf(M, s_m[wv]);
  float L = 0.f;
#pragma unroll
  for (int wv = 0; wv < 8; ++wv) L += s_l[wv] * __expf(s_m[wv] - M);
  const float factor = __expf(mw - M);               // this wave's rescale

  // ---- Phase B: weighted key accumulation (tile re-read hits L2/L0).
  const float* baseR = hr  + ((size_t)b * H_ + h0) * D_;
  const float* baseI = hgi + ((size_t)b * H_ + h0) * D_;
#pragma unroll 1
  for (int j = 0; j < 32; ++j) {
    const int d = j * 128 + lane * 4;
    const float* base = (d < D_) ? (baseR + d) : (baseI + (d - D_));
    float4 v = {0.f, 0.f, 0.f, 0.f};
#pragma unroll
    for (int r = 0; r < 16; ++r) {
      float4 kv = *(const float4*)(base + (size_t)r * D_);
      v.x += wr[r] * kv.x; v.y += wr[r] * kv.y;
      v.z += wr[r] * kv.z; v.w += wr[r] * kv.w;
    }
    atomicAdd(&s_acc[d + 0], factor * v.x);
    atomicAdd(&s_acc[d + 1], factor * v.y);
    atomicAdd(&s_acc[d + 2], factor * v.z);
    atomicAdd(&s_acc[d + 3], factor * v.w);
  }
  __syncthreads();

  float* p = partial + (size_t)blockIdx.x * PSTRIDE;
  if (tid == 0) { p[0] = M; p[1] = L; }
  for (int k = tid; k < TWO_D; k += 256) p[16 + k] = s_acc[k];
}

// ---------------------------------------------------------------- kernel 3
__global__ void __launch_bounds__(256)
combine_ema(const float* __restrict__ partial, const float* __restrict__ ema,
            const float* __restrict__ alpha_p, float* __restrict__ out) {
  const int b = blockIdx.x >> 4;
  const int d = ((blockIdx.x & 15) << 8) + threadIdx.x;
  const float* pb = partial + (size_t)(b * SEGS) * PSTRIDE;

  float m_s[SEGS];
  float M = -1e30f;
#pragma unroll
  for (int s = 0; s < SEGS; ++s) {
    m_s[s] = pb[(size_t)s * PSTRIDE];
    M = fmaxf(M, m_s[s]);
  }
  float L = 0.f, r = 0.f;
#pragma unroll
  for (int s = 0; s < SEGS; ++s) {
    float f = __expf(m_s[s] - M);
    L += f * pb[(size_t)s * PSTRIDE + 1];
    r += f * pb[(size_t)s * PSTRIDE + 16 + d];
  }
  float retrieved = r / L;
  float a = 1.0f / (1.0f + __expf(-alpha_p[0]));     // sigmoid(alpha)
  int g = b * TWO_D + d;
  out[g] = a * retrieved + (1.0f - a) * ema[g];
}

// ---------------------------------------------------------------- launcher
extern "C" void kernel_launch(void* const* d_in, const int* in_sizes, int n_in,
                              void* d_out, int out_size, void* d_ws, size_t ws_size,
                              hipStream_t stream) {
  const float* cr    = (const float*)d_in[0];   // current_state_real (B,D)
  const float* ci    = (const float*)d_in[1];   // current_state_imag (B,D)
  const float* hr    = (const float*)d_in[2];   // history_real (B,H,D)
  const float* hgi   = (const float*)d_in[3];   // history_imag (B,H,D)
  const float* ema   = (const float*)d_in[4];   // ema_state (B,2D)
  const float* wq    = (const float*)d_in[5];   // w_q (D)
  const float* bq    = (const float*)d_in[6];   // b_q (D)
  const float* alpha = (const float*)d_in[7];   // scalar
  const float* t     = (const float*)d_in[8];   // scalar
  float* out = (float*)d_out;

  // ws layout: [ q_bf16 : B*2D bf16 = 128KB ][ partials : 256*PSTRIDE f32 ~4.2MB ]
  __bf16* qbf = (__bf16*)d_ws;
  float* partial = (float*)((char*)d_ws + (size_t)B_ * TWO_D * sizeof(__bf16));

  build_query<<<(B_ * TWO_D) / 256, 256, 0, stream>>>(cr, ci, wq, bq, t, qbf);
  flash_partial<<<B_ * SEGS, 256, 0, stream>>>(hr, hgi, qbf, partial);
  combine_ema<<<B_ * SEGS, 256, 0, stream>>>(partial, ema, alpha, out);
}
